// GatedNNMF_88038239634058
// MI455X (gfx1250) — matrix operand order repaired and checked by reading, not compiled, not run on hardware
//
#include <hip/hip_runtime.h>

// ---------------------------------------------------------------------------
// GatedNNMF for MI455X (gfx1250): f16 WMMA (16x16x32) for the big GEMMs,
// f32 WMMA (16x16x4) for rank-64 NMF denominators and reconstruction.
// ---------------------------------------------------------------------------

#define B_DIM   8
#define S_DIM   1024
#define F_DIM   1024
#define FFN_DIM 4096
#define F2_DIM  2048
#define R_DIM   64
#define NMF_EPS 1e-6f
#define LNEPS   1e-5f

typedef __attribute__((ext_vector_type(16))) _Float16 v16h;
typedef __attribute__((ext_vector_type(8)))  _Float16 v8h;
typedef __attribute__((ext_vector_type(8)))  float    v8f;
typedef __attribute__((ext_vector_type(2)))  float    v2f;
typedef __attribute__((ext_vector_type(4)))  float    v4f;

#define WMMA_F16(a, b, c) \
  __builtin_amdgcn_wmma_f32_16x16x32_f16(false, (a), false, (b), (short)0, (c), false, false)
#define WMMA_F32(a, b, c) \
  __builtin_amdgcn_wmma_f32_16x16x4_f32(false, (a), false, (b), (short)0, (c), false, false)

// ---- fragment loaders -----------------------------------------------------
// f16 A frag 16x32 (ISA 7.12.2): lane L: M = L%16; lanes 0-15 hold K 0..7 /
// 16..23 in v0..7, lanes 16-31 hold K 8..15 / 24..31 -> two contiguous b128s.
static __device__ __forceinline__ v16h ldfragA_f16(const _Float16* A, int lda,
                                                   int m, int k0, int lane) {
  const int half = lane >> 4;
  const _Float16* p = A + (size_t)(m + (lane & 15)) * lda + k0 + 8 * half;
  v8h lo = *(const v8h*)(p);
  v8h hi = *(const v8h*)(p + 16);
  v16h r;
#pragma unroll
  for (int i = 0; i < 8; ++i) { r[i] = lo[i]; r[i + 8] = hi[i]; }
  return r;
}
// f16 B frag 32x16 with Bt stored N x K row-major: lane L: N = L%16;
// lanes 0-15: K 0..7 / 16..23, lanes 16-31: K 8..15 / 24..31 -> two b128s.
static __device__ __forceinline__ v16h ldfragB_f16(const _Float16* Bt, int ldb,
                                                   int n, int k0, int lane) {
  const int half = lane >> 4;
  const _Float16* p = Bt + (size_t)(n + (lane & 15)) * ldb + k0 + 8 * half;
  v8h lo = *(const v8h*)(p);
  v8h hi = *(const v8h*)(p + 16);
  v16h r;
#pragma unroll
  for (int i = 0; i < 8; ++i) { r[i] = lo[i]; r[i + 8] = hi[i]; }
  return r;
}
// f32 A frag 16x4: lane L: M = L%16, K = v + 2*(L/16) -> one b64.
static __device__ __forceinline__ v2f ldfragA_f32(const float* A, int lda,
                                                  int m, int k0, int lane) {
  const int half = lane >> 4;
  return *(const v2f*)(A + (size_t)(m + (lane & 15)) * lda + k0 + 2 * half);
}
// f32 B frag 4x16 from K-major matrix Bk (K x N), symmetric-safe usage.
static __device__ __forceinline__ v2f ldfragB_kmaj_f32(const float* Bk, int ldb,
                                                       int n, int k0, int lane) {
  const int half = lane >> 4;
  const int nn = n + (lane & 15);
  v2f r;
  r.x = Bk[(size_t)(k0 + 2 * half) * ldb + nn];
  r.y = Bk[(size_t)(k0 + 2 * half + 1) * ldb + nn];
  return r;
}

// ---- elementwise helpers --------------------------------------------------
static __device__ __forceinline__ float gelu_exact(float v) {
  return 0.5f * v * (1.0f + erff(v * 0.70710678118654752f));
}

// ---------------------------------------------------------------------------
// Generic f16 GEMM: C[b,m,n] = act( sum_k A[b,m,k]*Bt[b,n,k] + bias[n] )
// WG = 128 threads = 4 waves, wave tile 32x64, WG tile 128x64.
// Requires M%32==0 (or full-wave OOB), N%64==0, K%32==0.
// ---------------------------------------------------------------------------
__global__ __launch_bounds__(128) void gemm_f16_kernel(
    const _Float16* __restrict__ A, const _Float16* __restrict__ Bt,
    float* __restrict__ C, const float* __restrict__ bias,
    int M, int N, int K, long long sA, long long sB, long long sC, int act) {
  const int lane = threadIdx.x & 31;
  const int wv = threadIdx.x >> 5;
  const int bz = blockIdx.z;
  const _Float16* Ab = A + (size_t)bz * sA;
  const _Float16* Bb = Bt + (size_t)bz * sB;
  float* Cb = C + (size_t)bz * sC;
  const int m0 = blockIdx.x * 128 + wv * 32;
  const int n0 = blockIdx.y * 64;
  if (m0 >= M) return;

  v8f acc[2][4];
#pragma unroll
  for (int i = 0; i < 2; ++i)
#pragma unroll
    for (int j = 0; j < 4; ++j) acc[i][j] = (v8f){0.f,0.f,0.f,0.f,0.f,0.f,0.f,0.f};

  for (int k0 = 0; k0 < K; k0 += 32) {
    if (k0 + 32 < K) {  // -> global_prefetch_b8
      __builtin_prefetch(Ab + (size_t)(m0 + (lane & 15)) * K + k0 + 32, 0, 0);
      __builtin_prefetch(Bb + (size_t)(n0 + (lane & 15)) * K + k0 + 32, 0, 0);
    }
    v16h a0 = ldfragA_f16(Ab, K, m0, k0, lane);
    v16h a1 = ldfragA_f16(Ab, K, m0 + 16, k0, lane);
#pragma unroll
    for (int j = 0; j < 4; ++j) {
      v16h bf = ldfragB_f16(Bb, K, n0 + 16 * j, k0, lane);
      acc[0][j] = WMMA_F16(a0, bf, acc[0][j]);
      acc[1][j] = WMMA_F16(a1, bf, acc[1][j]);
    }
  }
  const int half = lane >> 4, nl = lane & 15;
#pragma unroll
  for (int i = 0; i < 2; ++i)
#pragma unroll
    for (int j = 0; j < 4; ++j) {
      const int n = n0 + 16 * j + nl;
      const float bb = bias ? bias[n] : 0.0f;
#pragma unroll
      for (int v = 0; v < 8; ++v) {
        const int m = m0 + 16 * i + v + 8 * half;
        float val = acc[i][j][v] + bb;
        if (act == 1) val = gelu_exact(val);
        Cb[(size_t)m * N + n] = val;
      }
    }
}

// ---------------------------------------------------------------------------
// Gram partials: Gp[b,chunk] += Xt(64xK chunk)^T... actually C = X^T X where
// Xt is the (64 x K) transposed-f16 copy. Deterministic K-split, no atomics.
// ---------------------------------------------------------------------------
__global__ __launch_bounds__(256) void gram_f16_kernel(
    const _Float16* __restrict__ Xt, float* __restrict__ Gpart,
    int K, int kc, long long sX) {
  const int lane = threadIdx.x & 31;
  const int wv = threadIdx.x >> 5;
  const int bz = blockIdx.z;
  const int nchunks = gridDim.x;
  const _Float16* Xb = Xt + (size_t)bz * sX;
  float* Gp = Gpart + ((size_t)bz * nchunks + blockIdx.x) * 4096;
  const int kbeg = blockIdx.x * kc;
  const int half = lane >> 4, nl = lane & 15;
#pragma unroll
  for (int t = 2 * wv; t < 2 * wv + 2; ++t) {
    const int m0 = (t >> 2) * 16, n0 = (t & 3) * 16;
    v8f acc = (v8f){0.f,0.f,0.f,0.f,0.f,0.f,0.f,0.f};
    for (int k = kbeg; k < kbeg + kc; k += 32) {
      v16h a = ldfragA_f16(Xb, K, m0, k, lane);
      v16h bf = ldfragB_f16(Xb, K, n0, k, lane);
      acc = WMMA_F16(a, bf, acc);
    }
#pragma unroll
    for (int v = 0; v < 8; ++v)
      Gp[(size_t)(m0 + v + 8 * half) * 64 + n0 + nl] = acc[v];
  }
}

__global__ void gram_reduce_kernel(const float* __restrict__ Gpart,
                                   float* __restrict__ G, int nchunks) {
  const int id = blockIdx.x * blockDim.x + threadIdx.x;  // B*4096
  const int b = id >> 12, e = id & 4095;
  float s = 0.f;
  for (int c = 0; c < nchunks; ++c)
    s += Gpart[((size_t)b * nchunks + c) * 4096 + e];
  G[(size_t)b * 4096 + e] = s;
}

// ---------------------------------------------------------------------------
// NMF multiplicative update: X = X * numer / (X @ G + eps), G (64x64, symm).
// Also emits transposed f16 copy XT (64 x M). f32 WMMA 16x16x4, K=64.
// In-place safe: each row's denominator reads only that row of X.
// ---------------------------------------------------------------------------
__global__ __launch_bounds__(256) void nmf_update_kernel(
    float* __restrict__ X, const float* __restrict__ numer,
    const float* __restrict__ G, _Float16* __restrict__ XT, int M) {
  const int lane = threadIdx.x & 31;
  const int wv = threadIdx.x >> 5;
  const int bz = blockIdx.z;
  float* Xb = X + (size_t)bz * M * 64;
  const float* Nb = numer + (size_t)bz * M * 64;
  const float* Gb = G + (size_t)bz * 4096;
  _Float16* Tb = XT + (size_t)bz * M * 64;  // (64, M)
  const int m0 = blockIdx.x * 128 + wv * 16;
  if (m0 >= M) return;
  v8f acc[4];
#pragma unroll
  for (int j = 0; j < 4; ++j) acc[j] = (v8f){0.f,0.f,0.f,0.f,0.f,0.f,0.f,0.f};
  for (int r0 = 0; r0 < 64; r0 += 4) {
    v2f a = ldfragA_f32(Xb, 64, m0, r0, lane);
#pragma unroll
    for (int j = 0; j < 4; ++j) {
      v2f bf = ldfragB_kmaj_f32(Gb, 64, 16 * j, r0, lane);
      acc[j] = WMMA_F32(a, bf, acc[j]);
    }
  }
  const int half = lane >> 4, nl = lane & 15;
#pragma unroll
  for (int j = 0; j < 4; ++j) {
    const int n = 16 * j + nl;
#pragma unroll
    for (int v = 0; v < 8; ++v) {
      const int m = m0 + v + 8 * half;
      const size_t idx = (size_t)m * 64 + n;
      const float val = Xb[idx] * Nb[idx] / (acc[j][v] + NMF_EPS);
      Xb[idx] = val;
      Tb[(size_t)n * M + m] = (_Float16)val;
    }
  }
}

// ---------------------------------------------------------------------------
// Reconstruction + gate: g[b,d,n] = f16( z1[b,d,n] * sum_r bas[b,d,r]*coef[b,n,r] )
// f32 WMMA, K = 64.  z1 = first half of h.
// ---------------------------------------------------------------------------
__global__ __launch_bounds__(256) void recon_gate_kernel(
    const float* __restrict__ bas, const float* __restrict__ coef,
    const float* __restrict__ h, _Float16* __restrict__ g) {
  const int lane = threadIdx.x & 31;
  const int wv = threadIdx.x >> 5;
  const int bz = blockIdx.z;
  const float* Bb = bas + (size_t)bz * S_DIM * R_DIM;
  const float* Cb = coef + (size_t)bz * F2_DIM * R_DIM;
  const float* Hb = h + (size_t)bz * S_DIM * FFN_DIM;
  _Float16* Gb = g + (size_t)bz * S_DIM * F2_DIM;
  const int m0 = blockIdx.x * 128 + wv * 16;
  const int n0 = blockIdx.y * 64;
  v8f acc[4];
#pragma unroll
  for (int j = 0; j < 4; ++j) acc[j] = (v8f){0.f,0.f,0.f,0.f,0.f,0.f,0.f,0.f};
  for (int r0 = 0; r0 < 64; r0 += 4) {
    v2f a = ldfragA_f32(Bb, 64, m0, r0, lane);
#pragma unroll
    for (int j = 0; j < 4; ++j) {
      v2f bf = ldfragA_f32(Cb, 64, n0 + 16 * j, r0, lane);  // N x K row-major
      acc[j] = WMMA_F32(a, bf, acc[j]);
    }
  }
  const int half = lane >> 4, nl = lane & 15;
#pragma unroll
  for (int j = 0; j < 4; ++j) {
    const int n = n0 + 16 * j + nl;
#pragma unroll
    for (int v = 0; v < 8; ++v) {
      const int m = m0 + v + 8 * half;
      const float z1 = Hb[(size_t)m * FFN_DIM + n];
      Gb[(size_t)m * F2_DIM + n] = (_Float16)(z1 * acc[j][v]);
    }
  }
}

// ---------------------------------------------------------------------------
// LayerNorm + ReLU over z2 (second half of h), output f16 xm.
// ---------------------------------------------------------------------------
__global__ __launch_bounds__(256) void ln_relu_kernel(
    const float* __restrict__ h, const float* __restrict__ gamma,
    const float* __restrict__ beta, _Float16* __restrict__ xm) {
  const int row = blockIdx.x;  // B*S
  const float* p = h + (size_t)row * FFN_DIM + F2_DIM;
  __shared__ float s1[256], s2[256];
  float vals[8];
  float sum = 0.f, sq = 0.f;
#pragma unroll
  for (int i = 0; i < 8; ++i) {
    vals[i] = p[threadIdx.x + 256 * i];
    sum += vals[i];
    sq += vals[i] * vals[i];
  }
  s1[threadIdx.x] = sum; s2[threadIdx.x] = sq;
  __syncthreads();
  for (int st = 128; st > 0; st >>= 1) {
    if (threadIdx.x < st) {
      s1[threadIdx.x] += s1[threadIdx.x + st];
      s2[threadIdx.x] += s2[threadIdx.x + st];
    }
    __syncthreads();
  }
  const float mu = s1[0] * (1.0f / F2_DIM);
  const float var = s2[0] * (1.0f / F2_DIM) - mu * mu;
  const float inv = rsqrtf(var + LNEPS);
#pragma unroll
  for (int i = 0; i < 8; ++i) {
    const int c = threadIdx.x + 256 * i;
    float y = (vals[i] - mu) * inv * gamma[c] + beta[c];
    y = fmaxf(y, 0.0f);
    xm[(size_t)row * F2_DIM + c] = (_Float16)y;
  }
}

// ---- softmax over R=64, also writes transposed f16 coef -------------------
__global__ void softmax_r_kernel(const float* __restrict__ numer,
                                 float* __restrict__ coef,
                                 _Float16* __restrict__ coefT) {
  const int id = blockIdx.x * blockDim.x + threadIdx.x;  // B * F2
  const int b = id >> 11, n = id & (F2_DIM - 1);
  const float* p = numer + (size_t)id * R_DIM;
  float mx = -3.4e38f;
  for (int r = 0; r < R_DIM; ++r) mx = fmaxf(mx, p[r]);
  float s = 0.f;
  for (int r = 0; r < R_DIM; ++r) s += expf(p[r] - mx);
  const float inv = 1.0f / s;
  for (int r = 0; r < R_DIM; ++r) {
    const float v = expf(p[r] - mx) * inv;
    coef[(size_t)id * R_DIM + r] = v;
    coefT[(size_t)b * R_DIM * F2_DIM + (size_t)r * F2_DIM + n] = (_Float16)v;
  }
}

// ---- tiled f16 transpose (rows x cols) -> (cols x rows), per batch --------
__global__ void transpose_f16_kernel(const _Float16* __restrict__ in,
                                     _Float16* __restrict__ out,
                                     int rows, int cols) {
  __shared__ _Float16 t[32][33];
  const int bz = blockIdx.z;
  const _Float16* inb = in + (size_t)bz * rows * cols;
  _Float16* outb = out + (size_t)bz * rows * cols;
  const int c0 = blockIdx.x * 32, r0 = blockIdx.y * 32;
  const int tx = threadIdx.x, ty = threadIdx.y;  // (32, 8)
#pragma unroll
  for (int i = 0; i < 4; ++i)
    t[ty + 8 * i][tx] = inb[(size_t)(r0 + ty + 8 * i) * cols + c0 + tx];
  __syncthreads();
#pragma unroll
  for (int i = 0; i < 4; ++i)
    outb[(size_t)(c0 + ty + 8 * i) * rows + r0 + tx] = t[tx][ty + 8 * i];
}

// ---- bases init: broadcast over batch + transposed f16 copy ---------------
__global__ void init_bases_kernel(const float* __restrict__ bases,
                                  float* __restrict__ bas,
                                  _Float16* __restrict__ basT) {
  const int idx = blockIdx.x * blockDim.x + threadIdx.x;  // S*R
  const int b = blockIdx.y;
  const int d = idx >> 6, r = idx & 63;
  const float v = bases[idx];
  bas[(size_t)b * S_DIM * R_DIM + idx] = v;
  basT[(size_t)b * S_DIM * R_DIM + (size_t)r * S_DIM + d] = (_Float16)v;
}

// ---- f32 -> f16 conversion (4-wide) ---------------------------------------
__global__ void cvt_f16_kernel(const float* __restrict__ in,
                               _Float16* __restrict__ out, int n4) {
  const int i = blockIdx.x * blockDim.x + threadIdx.x;
  if (i < n4) {
    v4f v = *(const v4f*)(in + (size_t)i * 4);
    out[(size_t)i * 4 + 0] = (_Float16)v.x;
    out[(size_t)i * 4 + 1] = (_Float16)v.y;
    out[(size_t)i * 4 + 2] = (_Float16)v.z;
    out[(size_t)i * 4 + 3] = (_Float16)v.w;
  }
}

// ---------------------------------------------------------------------------
extern "C" void kernel_launch(void* const* d_in, const int* in_sizes, int n_in,
                              void* d_out, int out_size, void* d_ws, size_t ws_size,
                              hipStream_t stream) {
  (void)in_sizes; (void)n_in; (void)out_size; (void)ws_size;
  const float* x     = (const float*)d_in[0];
  const float* Wu    = (const float*)d_in[1];
  const float* bu    = (const float*)d_in[2];
  const float* gamma = (const float*)d_in[3];
  const float* beta  = (const float*)d_in[4];
  const float* Wv    = (const float*)d_in[5];
  const float* bv    = (const float*)d_in[6];
  const float* bases = (const float*)d_in[7];
  float* out = (float*)d_out;

  char* ws = (char*)d_ws;
  size_t off = 0;
  auto alloc = [&](size_t bytes) -> void* {
    void* p = ws + off;
    off += (bytes + 255) & ~(size_t)255;
    return p;
  };
  const long long MS = (long long)B_DIM * S_DIM;  // 8192
  _Float16* x_h    = (_Float16*)alloc(MS * F_DIM * 2);
  _Float16* Wu_h   = (_Float16*)alloc((size_t)FFN_DIM * F_DIM * 2);
  _Float16* Wv_h   = (_Float16*)alloc((size_t)F_DIM * F2_DIM * 2);
  float*    h      = (float*)alloc(MS * FFN_DIM * 4);
  _Float16* xm_h   = (_Float16*)alloc(MS * F2_DIM * 2);
  _Float16* xmT_h  = (_Float16*)alloc(MS * F2_DIM * 2);
  float*    basf   = (float*)alloc((size_t)B_DIM * S_DIM * R_DIM * 4);
  _Float16* basT_h = (_Float16*)alloc((size_t)B_DIM * S_DIM * R_DIM * 2);
  float*    coef   = (float*)alloc((size_t)B_DIM * F2_DIM * R_DIM * 4);
  _Float16* coefT_h= (_Float16*)alloc((size_t)B_DIM * F2_DIM * R_DIM * 2);
  float*    numC   = (float*)alloc((size_t)B_DIM * F2_DIM * R_DIM * 4);
  float*    numB   = (float*)alloc((size_t)B_DIM * S_DIM * R_DIM * 4);
  float*    G      = (float*)alloc((size_t)B_DIM * 4096 * 4);
  float*    Gpart  = (float*)alloc((size_t)B_DIM * 8 * 4096 * 4);
  _Float16* g_h    = (_Float16*)alloc(MS * F2_DIM * 2);

  auto gemm = [&](const _Float16* A, const _Float16* Bt, float* C,
                  const float* bias, int M, int N, int K,
                  long long sA, long long sB, long long sC, int act, int batch) {
    dim3 grid(M / 128, N / 64, batch);
    gemm_f16_kernel<<<grid, 128, 0, stream>>>(A, Bt, C, bias, M, N, K, sA, sB, sC, act);
  };

  // --- conversions ---
  cvt_f16_kernel<<<(MS * F_DIM / 4 + 255) / 256, 256, 0, stream>>>(x, x_h, MS * F_DIM / 4);
  cvt_f16_kernel<<<(FFN_DIM * F_DIM / 4 + 255) / 256, 256, 0, stream>>>(Wu, Wu_h, FFN_DIM * F_DIM / 4);
  cvt_f16_kernel<<<(F_DIM * F2_DIM / 4 + 255) / 256, 256, 0, stream>>>(Wv, Wv_h, F_DIM * F2_DIM / 4);

  // --- GEMM1: h = GELU(x @ Wu^T + bu) ---
  gemm(x_h, Wu_h, h, bu, (int)MS, FFN_DIM, F_DIM, 0, 0, 0, /*gelu*/1, 1);

  // --- LN + ReLU -> xm (f16), then transpose -> xmT ---
  ln_relu_kernel<<<(int)MS, 256, 0, stream>>>(h, gamma, beta, xm_h);
  transpose_f16_kernel<<<dim3(F2_DIM / 32, S_DIM / 32, B_DIM), dim3(32, 8), 0, stream>>>(
      xm_h, xmT_h, S_DIM, F2_DIM);

  // --- bases: broadcast + transposed f16 ---
  init_bases_kernel<<<dim3(S_DIM * R_DIM / 256, B_DIM), 256, 0, stream>>>(bases, basf, basT_h);

  const long long sXmT = (long long)F2_DIM * F_DIM;   // per-batch xmT (2048 x 1024)
  const long long sXm  = (long long)S_DIM * F2_DIM;   // per-batch xm  (1024 x 2048)
  const long long sBT  = (long long)R_DIM * S_DIM;    // basT (64 x 1024)
  const long long sCT  = (long long)R_DIM * F2_DIM;   // coefT (64 x 2048)
  const long long sNC  = (long long)F2_DIM * R_DIM;
  const long long sNB  = (long long)S_DIM * R_DIM;

  // --- init coef = softmax_R(xm^T @ bas) ---
  gemm(xmT_h, basT_h, numC, nullptr, F2_DIM, R_DIM, S_DIM, sXmT, sBT, sNC, 0, B_DIM);
  softmax_r_kernel<<<B_DIM * F2_DIM / 256, 256, 0, stream>>>(numC, coef, coefT_h);

  // --- 6 NMF iterations + final compute_coef (it==6 does coef update only) ---
  for (int it = 0; it <= 6; ++it) {
    // numerC = xm^T @ bas ; G = bas^T bas ; coef *= numerC / (coef@G + eps)
    gemm(xmT_h, basT_h, numC, nullptr, F2_DIM, R_DIM, S_DIM, sXmT, sBT, sNC, 0, B_DIM);
    gram_f16_kernel<<<dim3(4, 1, B_DIM), 256, 0, stream>>>(basT_h, Gpart, S_DIM, 256, sBT);
    gram_reduce_kernel<<<B_DIM * 4096 / 256, 256, 0, stream>>>(Gpart, G, 4);
    nmf_update_kernel<<<dim3(F2_DIM / 128, 1, B_DIM), 256, 0, stream>>>(coef, numC, G, coefT_h, F2_DIM);
    if (it == 6) break;
    // numerB = xm @ coef ; G2 = coef^T coef ; bas *= numerB / (bas@G2 + eps)
    gemm(xm_h, coefT_h, numB, nullptr, S_DIM, R_DIM, F2_DIM, sXm, sCT, sNB, 0, B_DIM);
    gram_f16_kernel<<<dim3(8, 1, B_DIM), 256, 0, stream>>>(coefT_h, Gpart, F2_DIM, 256, sCT);
    gram_reduce_kernel<<<B_DIM * 4096 / 256, 256, 0, stream>>>(Gpart, G, 8);
    nmf_update_kernel<<<dim3(S_DIM / 128, 1, B_DIM), 256, 0, stream>>>(basf, numB, G, basT_h, S_DIM);
  }

  // --- g = f16(z1 * (bas @ coef^T)) ---
  recon_gate_kernel<<<dim3(S_DIM / 128, F2_DIM / 64, B_DIM), 256, 0, stream>>>(basf, coef, h, g_h);

  // --- GEMM3: out = g @ Wv^T + bv ---
  gemm(g_h, Wv_h, out, bv, (int)MS, F_DIM, F2_DIM, 0, 0, 0, 0, 1);
}
